// EfficientSlidingWindowMultiheadAttention_32332513804975
// MI455X (gfx1250) — matrix operand order repaired and checked by reading, not compiled, not run on hardware
//
#include <hip/hip_runtime.h>

typedef _Float16 h16;
typedef __attribute__((ext_vector_type(16))) _Float16 v16h;
typedef __attribute__((ext_vector_type(8)))  _Float16 v8h;
typedef __attribute__((ext_vector_type(8)))  float    v8f;

typedef int v4i32 __attribute__((vector_size(16)));
typedef __attribute__((address_space(1))) v4i32 gv4i32;
typedef __attribute__((address_space(3))) v4i32 lv4i32;

#define CB 2
#define CS 1024
#define CH 1024
#define CNH 16
#define CHD 64
#define CW 64

union AFrag { v16h v; v8h h[2]; };
struct Frags { AFrag a[2]; v16h b[4]; };

static __device__ inline v8f zero8f() {
    v8f z;
#pragma unroll
    for (int i = 0; i < 8; ++i) z[i] = 0.0f;
    return z;
}
static __device__ inline v16h zero16h() {
    v16h z;
#pragma unroll
    for (int i = 0; i < 16; ++i) z[i] = (h16)0.0f;
    return z;
}

// ---- gfx1250 async global->LDS copy (16B per lane) + wait ----
static __device__ inline void async_copy_b128(const h16* gsrc, h16* ldst) {
#if __has_builtin(__builtin_amdgcn_global_load_async_to_lds_b128)
    __builtin_amdgcn_global_load_async_to_lds_b128(
        (gv4i32*)(uintptr_t)gsrc,
        (lv4i32*)(unsigned)(uintptr_t)ldst,
        0, 0);
#else
    unsigned loff = (unsigned)(uintptr_t)ldst;
    unsigned long long ga = (unsigned long long)(uintptr_t)gsrc;
    asm volatile("global_load_async_to_lds_b128 %0, %1, off"
                 :: "v"(loff), "v"(ga) : "memory");
#endif
}
static __device__ inline void wait_async0() {
#if __has_builtin(__builtin_amdgcn_s_wait_asynccnt)
    __builtin_amdgcn_s_wait_asynccnt(0);
#else
    asm volatile("s_wait_asynccnt 0x0" ::: "memory");
#endif
}

// ---------------- f32 -> f16 convert ----------------
__global__ void k_cvt(const float* __restrict__ src, h16* __restrict__ dst, int n) {
    int i = blockIdx.x * blockDim.x + threadIdx.x;
    if (i < n) dst[i] = (h16)src[i];
}

// ---- fragment loaders ----
static __device__ inline void load_frags(Frags& f, const h16* lA,
                                         const h16* const wrow[4],
                                         int ln, int half, int k0) {
#pragma unroll
    for (int s = 0; s < 2; ++s) {
        const h16* r = lA + (size_t)(s * 16 + ln) * CH + k0 + half * 8;
        f.a[s].h[0] = *(const v8h*)(r);
        f.a[s].h[1] = *(const v8h*)(r + 16);
    }
#pragma unroll
    for (int t = 0; t < 4; ++t) {
        v8h b0 = *(const v8h*)(wrow[t] + k0);
        v8h b1 = *(const v8h*)(wrow[t] + k0 + 16);
        f.b[t] = __builtin_shufflevector(b0, b1, 0,1,2,3,4,5,6,7,8,9,10,11,12,13,14,15);
        if (k0 + 64 < CH) __builtin_prefetch(wrow[t] + k0 + 64, 0, 3);
    }
}
static __device__ inline void mma_frags(v8f acc[2][4], const Frags& f) {
#pragma unroll
    for (int s = 0; s < 2; ++s)
#pragma unroll
        for (int t = 0; t < 4; ++t)
            acc[s][t] = __builtin_amdgcn_wmma_f32_16x16x32_f16(
                false, f.a[s].v, false, f.b[t], (short)0, acc[s][t], false, false);
}

// ---------------- QKV GEMM + bias + RoPE ----------------
// out[m, n] = sum_k x[m,k] * Wqkv[n,k] + bqkv[n], then RoPE on q/k sections.
// One wave computes a 32x64 tile. A tile (32x1024) staged in LDS per block.
__global__ void __launch_bounds__(256) k_qkv(
        const h16* __restrict__ x16, const h16* __restrict__ w16,
        const float* __restrict__ bqkv, const float* __restrict__ angles,
        h16* __restrict__ qkv16) {
    __shared__ __align__(16) h16 lA[32 * CH];   // 64 KB

    const int wid  = blockIdx.x * 8 + (threadIdx.x >> 5);
    const int lane = threadIdx.x & 31;
    const int half = lane >> 4;
    const int ln   = lane & 15;

    const int mt = wid / 48;        // 64 M-tiles of 32 rows (uniform per block)
    const int nt = wid % 48;        // 48 N-tiles of 64 cols
    const int m0 = mt * 32;
    const int n0 = nt * 64;

    // async-stage A tile (contiguous 32*1024 f16) into LDS
    {
        const h16* gA = x16 + (size_t)m0 * CH;
        for (int c = threadIdx.x; c < 8192; c += 256)
            async_copy_b128(gA + c * 8, &lA[c * 8]);
        wait_async0();
        __syncthreads();
    }

    const h16* wrow[4];
#pragma unroll
    for (int t = 0; t < 4; ++t)
        wrow[t] = w16 + (size_t)(n0 + t * 16 + ln) * CH + half * 8;

    v8f acc[2][4];
#pragma unroll
    for (int s = 0; s < 2; ++s)
#pragma unroll
        for (int t = 0; t < 4; ++t) acc[s][t] = zero8f();

    // copy-free ping-pong pipeline: 32 K-steps of 32
    Frags fA, fB;
    load_frags(fA, lA, wrow, ln, half, 0);
    for (int j = 0; j < 15; ++j) {
        load_frags(fB, lA, wrow, ln, half, (2 * j + 1) * 32);
        mma_frags(acc, fA);
        load_frags(fA, lA, wrow, ln, half, (2 * j + 2) * 32);
        mma_frags(acc, fB);
    }
    load_frags(fB, lA, wrow, ln, half, 31 * 32);
    mma_frags(acc, fA);
    mma_frags(acc, fB);

    // epilogue: bias + RoPE + f16 store
#pragma unroll
    for (int t = 0; t < 4; ++t) {
        const int n = n0 + t * 16 + ln;
        const float bias = bqkv[n];
        const int j192 = n % 192;
        const bool isqk = (j192 < 128);
        const int pidx = (j192 & 63) >> 1;
        const bool even = ((n & 1) == 0);
#pragma unroll
        for (int s = 0; s < 2; ++s) {
#pragma unroll
            for (int j = 0; j < 8; ++j) {
                const int mg = m0 + s * 16 + half * 8 + j;
                const int b  = mg >> 10;
                const int sq = mg & (CS - 1);
                float val = acc[s][t][j] + bias;
                float sn = 0.0f, cs = 1.0f;
                if (isqk) {
                    float ang = angles[sq * 32 + pidx];
                    __sincosf(ang, &sn, &cs);
                }
                float partner = __shfl_xor(val, 1, 32);
                float r = isqk ? (even ? val * cs - partner * sn
                                       : partner * sn + val * cs)
                               : val;
                qkv16[(size_t)(b * CS + sq) * 3072 + n] = (h16)r;
            }
        }
    }
}

// ---------------- Sliding-window attention ----------------
// One wave per (b, head, 16-query tile). 4 waves / block.
__global__ void k_attn(const h16* __restrict__ qkv16, h16* __restrict__ ctx16) {
    __shared__ __align__(32) h16 lAttn[4][16 * 96];

    const int w    = threadIdx.x >> 5;
    const int lane = threadIdx.x & 31;
    const int half = lane >> 4;
    const int ln   = lane & 15;

    const int wt  = blockIdx.x * 4 + w;            // 2048 tiles
    const int b   = wt >> 10;
    const int rem = wt & 1023;
    const int nh  = rem >> 6;
    const int s0  = (rem & 63) << 4;

    const h16* qbase = qkv16 + (size_t)(b * CS) * 3072 + nh * 192;

    AFrag aQ[2];
#pragma unroll
    for (int kk = 0; kk < 2; ++kk) {
        const h16* qr = qbase + (size_t)(s0 + ln) * 3072 + kk * 32;
        aQ[kk].h[0] = *(const v8h*)(qr + half * 8);
        aQ[kk].h[1] = *(const v8h*)(qr + 16 + half * 8);
    }

    v8f sc[5];
#pragma unroll
    for (int t = 0; t < 5; ++t) sc[t] = zero8f();

#pragma unroll
    for (int t = 0; t < 5; ++t) {
        const int p = s0 - 32 + t * 16 + ln;
        const bool okp = (p >= 0) && (p < CS);
#pragma unroll
        for (int kk = 0; kk < 2; ++kk) {
            v16h bk = zero16h();
            if (okp) {
                const h16* kr = qbase + (size_t)p * 3072 + 64 + kk * 32 + half * 8;
                v8h k0v = *(const v8h*)(kr);
                v8h k1v = *(const v8h*)(kr + 16);
                bk = __builtin_shufflevector(k0v, k1v, 0,1,2,3,4,5,6,7,8,9,10,11,12,13,14,15);
            }
            sc[t] = __builtin_amdgcn_wmma_f32_16x16x32_f16(
                false, aQ[kk].v, false, bk, (short)0, sc[t], false, false);
        }
    }

    for (int e = lane; e < 256; e += 32)
        lAttn[w][(e >> 4) * 96 + 80 + (e & 15)] = (h16)0.0f;

    const float scale = 0.125f;
#pragma unroll
    for (int j = 0; j < 8; ++j) {
        const int row = half * 8 + j;
        float mx = -3.4e38f;
#pragma unroll
        for (int t = 0; t < 5; ++t) {
            const int d = t * 16 + ln - row;
            const bool ok = (d >= 0) && (d < CW);
            float v = ok ? sc[t][j] * scale : -3.4e38f;
            mx = fmaxf(mx, v);
        }
#pragma unroll
        for (int off = 1; off < 16; off <<= 1)
            mx = fmaxf(mx, __shfl_xor(mx, off, 32));

        float ev[5];
        float sm = 0.0f;
#pragma unroll
        for (int t = 0; t < 5; ++t) {
            const int d = t * 16 + ln - row;
            const bool ok = (d >= 0) && (d < CW);
            float e = ok ? __expf(sc[t][j] * scale - mx) : 0.0f;
            ev[t] = e;
            sm += e;
        }
#pragma unroll
        for (int off = 1; off < 16; off <<= 1)
            sm += __shfl_xor(sm, off, 32);

        const float inv = 1.0f / sm;
#pragma unroll
        for (int t = 0; t < 5; ++t)
            lAttn[w][row * 96 + t * 16 + ln] = (h16)(ev[t] * inv);
    }

    v8f cc[4];
#pragma unroll
    for (int nt = 0; nt < 4; ++nt) cc[nt] = zero8f();

#pragma unroll
    for (int ks = 0; ks < 3; ++ks) {
        AFrag aA;
        aA.h[0] = *(const v8h*)(&lAttn[w][ln * 96 + ks * 32 + half * 8]);
        aA.h[1] = *(const v8h*)(&lAttn[w][ln * 96 + ks * 32 + 16 + half * 8]);
#pragma unroll
        for (int nt = 0; nt < 4; ++nt) {
            v16h bv;
#pragma unroll
            for (int i = 0; i < 16; ++i) {
                const int kg = ks * 32 + (i < 8 ? half * 8 + i : 16 + half * 8 + (i - 8));
                const int p  = s0 - 32 + kg;
                h16 vv = (h16)0.0f;
                if (kg < 80 && p >= 0 && p < CS)
                    vv = qbase[(size_t)p * 3072 + 128 + nt * 16 + ln];
                bv[i] = vv;
            }
            cc[nt] = __builtin_amdgcn_wmma_f32_16x16x32_f16(
                false, aA.v, false, bv, (short)0, cc[nt], false, false);
        }
    }

#pragma unroll
    for (int nt = 0; nt < 4; ++nt) {
#pragma unroll
        for (int j = 0; j < 8; ++j) {
            const int s  = s0 + half * 8 + j;
            const int hd = nt * 16 + ln;
            ctx16[(size_t)(b * CS + s) * CH + nh * CHD + hd] = (h16)cc[nt][j];
        }
    }
}

// ---------------- Output GEMM + bias ----------------
// One wave computes a 32x64 tile. A tile (32x1024) staged in LDS per block.
__global__ void __launch_bounds__(256) k_out(
        const h16* __restrict__ ctx16, const h16* __restrict__ w16,
        const float* __restrict__ bout, float* __restrict__ out) {
    __shared__ __align__(16) h16 lA[32 * CH];   // 64 KB

    const int wid  = blockIdx.x * 8 + (threadIdx.x >> 5);
    const int lane = threadIdx.x & 31;
    const int half = lane >> 4;
    const int ln   = lane & 15;

    const int mt = wid / 16;        // 64 M-tiles (uniform per block)
    const int nt = wid % 16;        // 16 N-tiles of 64
    const int m0 = mt * 32;
    const int n0 = nt * 64;

    {
        const h16* gA = ctx16 + (size_t)m0 * CH;
        for (int c = threadIdx.x; c < 8192; c += 256)
            async_copy_b128(gA + c * 8, &lA[c * 8]);
        wait_async0();
        __syncthreads();
    }

    const h16* wrow[4];
#pragma unroll
    for (int t = 0; t < 4; ++t)
        wrow[t] = w16 + (size_t)(n0 + t * 16 + ln) * CH + half * 8;

    v8f acc[2][4];
#pragma unroll
    for (int s = 0; s < 2; ++s)
#pragma unroll
        for (int t = 0; t < 4; ++t) acc[s][t] = zero8f();

    Frags fA, fB;
    load_frags(fA, lA, wrow, ln, half, 0);
    for (int j = 0; j < 15; ++j) {
        load_frags(fB, lA, wrow, ln, half, (2 * j + 1) * 32);
        mma_frags(acc, fA);
        load_frags(fA, lA, wrow, ln, half, (2 * j + 2) * 32);
        mma_frags(acc, fB);
    }
    load_frags(fB, lA, wrow, ln, half, 31 * 32);
    mma_frags(acc, fA);
    mma_frags(acc, fB);

#pragma unroll
    for (int t = 0; t < 4; ++t) {
        const int n = n0 + t * 16 + ln;
        const float bias = bout[n];
#pragma unroll
        for (int s = 0; s < 2; ++s) {
#pragma unroll
            for (int j = 0; j < 8; ++j) {
                const int m = m0 + s * 16 + half * 8 + j;
                out[(size_t)m * CH + n] = acc[s][t][j] + bias;
            }
        }
    }
}

extern "C" void kernel_launch(void* const* d_in, const int* in_sizes, int n_in,
                              void* d_out, int out_size, void* d_ws, size_t ws_size,
                              hipStream_t stream) {
    (void)in_sizes; (void)n_in; (void)out_size; (void)ws_size;

    const float* x      = (const float*)d_in[0];   // (2,1024,1024)
    const float* Wqkv   = (const float*)d_in[1];   // (3072,1024)
    const float* bqkv   = (const float*)d_in[2];   // (3072,)
    const float* Wout   = (const float*)d_in[3];   // (1024,1024)
    const float* bout   = (const float*)d_in[4];   // (1024,)
    const float* angles = (const float*)d_in[5];   // (1024,32)
    float* out = (float*)d_out;

    char* ws = (char*)d_ws;
    h16* x16    = (h16*)(ws);                                   // 2048*1024
    h16* wqkv16 = (h16*)(ws + 4194304);                         // 3072*1024
    h16* wout16 = (h16*)(ws + 4194304 + 6291456);               // 1024*1024
    h16* qkv16  = (h16*)(ws + 4194304 + 6291456 + 2097152);     // 2048*3072
    h16* ctx16  = (h16*)(ws + 4194304 + 6291456 + 2097152 + 12582912); // 2048*1024

    const int nx = CB * CS * CH;
    const int nw = 3 * CH * CH;
    const int no = CH * CH;

    k_cvt<<<(nx + 255) / 256, 256, 0, stream>>>(x, x16, nx);
    k_cvt<<<(nw + 255) / 256, 256, 0, stream>>>(Wqkv, wqkv16, nw);
    k_cvt<<<(no + 255) / 256, 256, 0, stream>>>(Wout, wout16, no);

    // 64 M-tiles(32) * 48 N-tiles(64) = 3072 waves, 8 per 256-thread block
    k_qkv<<<384, 256, 0, stream>>>(x16, wqkv16, bqkv, angles, qkv16);

    // 2 * 16 * 64 = 2048 (b, head, query-tile) waves, 4 per 128-thread block
    k_attn<<<512, 128, 0, stream>>>(qkv16, ctx16);

    // 64 M-tiles(32) * 16 N-tiles(64) = 1024 waves, 8 per 256-thread block
    k_out<<<128, 256, 0, stream>>>(ctx16, wout16, bout, out);
}
